// GridificationLayer_30674656428245
// MI455X (gfx1250) — matrix operand order repaired and checked by reading, not compiled, not run on hardware
//
#include <hip/hip_runtime.h>

typedef __attribute__((ext_vector_type(16))) __bf16 v16bf;
typedef __attribute__((ext_vector_type(8)))  __bf16 v8bf;
typedef __attribute__((ext_vector_type(8)))  float  v8f;

#define H128 128

__device__ __forceinline__ v8f wmma_bf16(v16bf a, v16bf b, v8f c) {
  // D = A(16x32 bf16) * B(32x16 bf16) + C(16x16 f32)
  return __builtin_amdgcn_wmma_f32_16x16x32_bf16(
      /*neg_a=*/false, a, /*neg_b=*/false, b,
      /*c_mod=*/(short)0, c, /*reuse_a=*/false, /*reuse_b=*/false);
}

__device__ __forceinline__ float silu_f(float x) {
  return x / (1.0f + __expf(-x));
}

// CDNA5 async global->LDS copy of 16 bytes (GLOBAL_LOAD_ASYNC_TO_LDS_B128,
// tracked by ASYNCcnt). lds_off is the 32-bit LDS byte offset (low 32 bits of
// the generic shared pointer per the flat->LDS aperture truncation rule).
__device__ __forceinline__ void async_copy_b128(unsigned lds_off, const void* gsrc) {
  asm volatile("global_load_async_to_lds_b128 %0, %1, off"
               :: "v"(lds_off), "v"((unsigned long long)(size_t)gsrc)
               : "memory");
}

__device__ __forceinline__ void wait_async0() {
#if __has_builtin(__builtin_amdgcn_s_wait_asynccnt)
  __builtin_amdgcn_s_wait_asynccnt(0);
#else
  asm volatile("s_wait_asynccnt 0x0" ::: "memory");
#endif
}

// A fragment (16x32 bf16) from a bf16 row in LDS.
// Lane layout: M = lane&15, khalf = (lane>>4)*8.
// e[0..7]  = A[M][kA + 0..7]      (kA = ks*32 + khalf)
// e[8..15] = A[M][kA + 16 + 0..7]
__device__ __forceinline__ v16bf a_from_bf16row(const __bf16* row, int kA) {
  v8bf lo = *(const v8bf*)(row + kA);
  v8bf hi = *(const v8bf*)(row + kA + 16);
  v16bf a;
#pragma unroll
  for (int i = 0; i < 8; ++i) { a[i] = lo[i]; a[i + 8] = hi[i]; }
  return a;
}

// A fragment from an f32 row in global memory, with an optional scale
// (used to fold the scatter-mean divide into the update-MLP input load).
__device__ __forceinline__ v16bf a_from_f32row(const float* row, int kA, float scale) {
  v16bf a;
#pragma unroll
  for (int i = 0; i < 8; ++i) a[i] = (__bf16)(row[kA + i] * scale);
#pragma unroll
  for (int i = 0; i < 8; ++i) a[i + 8] = (__bf16)(row[kA + 16 + i] * scale);
  return a;
}

// Pre-packed B fragment: one contiguous 32-byte load per lane.
__device__ __forceinline__ v16bf b_frag(const __bf16* Wp, int wave, int ksteps,
                                        int ks, int lane) {
  return *(const v16bf*)(Wp + (((size_t)wave * ksteps + ks) * 32 + lane) * 16);
}

// ---------------------------------------------------------------------------
// Pack a row-major [K x 128] f32 weight into bf16 B-fragment layout.
// Fragment t = (nt*ksteps + ks)*32 + lane holds 16 bf16:
//   n = (lane&15) + 16*nt ; k = ks*32 + ((lane>>4)<<3) + (i<8 ? i : i+8)
// k >= kvalid is zero-filled (pads em_w1's K=6 up to 32).
// ---------------------------------------------------------------------------
__global__ void pack_w_kernel(const float* __restrict__ W, __bf16* __restrict__ P,
                              int ksteps, int kvalid) {
  int t = blockIdx.x * blockDim.x + threadIdx.x;
  int total = 8 * ksteps * 32;
  if (t >= total) return;
  int lane = t & 31;
  int ksnt = t >> 5;
  int ks = ksnt % ksteps;
  int nt = ksnt / ksteps;
  int n = (lane & 15) + 16 * nt;
  int kbase = ks * 32 + ((lane >> 4) << 3);
  __bf16* dst = P + (size_t)t * 16;
#pragma unroll
  for (int i = 0; i < 16; ++i) {
    int k = kbase + (i < 8 ? i : i + 8);
    dst[i] = (__bf16)((k < kvalid) ? W[(size_t)k * H128 + n] : 0.0f);
  }
}

__global__ void zero_f32_kernel(float* __restrict__ p, long long n) {
  long long i = (long long)blockIdx.x * blockDim.x + threadIdx.x;
  if (i < n) p[i] = 0.0f;
}

// ---------------------------------------------------------------------------
// Fused 128 -> 128 (SiLU) -> 128 MLP. One block = 16 rows x 128 cols.
// 8 waves; wave w owns output columns [16w, 16w+16).
//  - cnts != nullptr: input row scaled by 1/max(cnts[row],1) (scatter-mean).
//  - Yb != nullptr: bf16 output (node MLP intermediate), else f32 to Yf.
// ---------------------------------------------------------------------------
__global__ __launch_bounds__(256) void mlp128_kernel(
    const float* __restrict__ X, const float* __restrict__ cnts,
    const __bf16* __restrict__ W1p, const float* __restrict__ b1,
    const __bf16* __restrict__ W2p, const float* __restrict__ b2,
    float* __restrict__ Yf, __bf16* __restrict__ Yb, int nrows) {
  __shared__ __bf16 sH1[16][H128];

  const int wave  = threadIdx.x >> 5;
  const int lane  = threadIdx.x & 31;
  const int half  = lane >> 4;
  const int nn    = (lane & 15) + wave * 16;
  const int khalf = half * 8;
  const int row0  = blockIdx.x * 16;

  int M = row0 + (lane & 15);
  if (M >= nrows) M = nrows - 1;
  const float* rowp = X + (size_t)M * H128;
  float scale = 1.0f;
  if (cnts) scale = 1.0f / fmaxf(cnts[M], 1.0f);

  // Layer 1: X @ W1
  v8f acc = {};
#pragma unroll
  for (int ks = 0; ks < 4; ++ks) {
    v16bf a = a_from_f32row(rowp, ks * 32 + khalf, scale);
    v16bf b = b_frag(W1p, wave, 4, ks, lane);
    acc = wmma_bf16(a, b, acc);
  }
  const float bias1 = b1[nn];
#pragma unroll
  for (int r = 0; r < 8; ++r)
    sH1[r + half * 8][nn] = (__bf16)silu_f(acc[r] + bias1);
  __syncthreads();

  // Layer 2: H1 @ W2 (no activation)
  v8f acc2 = {};
  const __bf16* hrow = &sH1[lane & 15][0];
#pragma unroll
  for (int ks = 0; ks < 4; ++ks) {
    v16bf a = a_from_bf16row(hrow, ks * 32 + khalf);
    v16bf b = b_frag(W2p, wave, 4, ks, lane);
    acc2 = wmma_bf16(a, b, acc2);
  }
  const float bias2 = b2[nn];
  if (Yb) {
#pragma unroll
    for (int r = 0; r < 8; ++r) {
      int Mo = row0 + half * 8 + r;
      if (Mo < nrows) Yb[(size_t)Mo * H128 + nn] = (__bf16)(acc2[r] + bias2);
    }
  } else {
#pragma unroll
    for (int r = 0; r < 8; ++r) {
      int Mo = row0 + half * 8 + r;
      if (Mo < nrows) Yf[(size_t)Mo * H128 + nn] = acc2[r] + bias2;
    }
  }
}

// ---------------------------------------------------------------------------
// Per-16-edge tile: edge geometry -> edge MLP -> concat(h_nodes[src], edge)
// -> message MLP -> f32 atomic scatter-add into sums/counts.
// h_nodes gather uses GLOBAL_LOAD_ASYNC_TO_LDS_B128 (raw bf16 rows).
// 17 WMMAs per wave, 8 waves per block.
// ---------------------------------------------------------------------------
__global__ __launch_bounds__(256) void edge_msg_kernel(
    const __bf16* __restrict__ h_nodes, const float* __restrict__ node_pos,
    const float* __restrict__ grid_pos, const int* __restrict__ edge_index,
    int E_total, const float* __restrict__ orient,
    const __bf16* __restrict__ em_w1p, const float* __restrict__ em_b1,
    const __bf16* __restrict__ em_w2p, const float* __restrict__ em_b2,
    const __bf16* __restrict__ mm_w1p, const float* __restrict__ mm_b1,
    const __bf16* __restrict__ mm_w2p, const float* __restrict__ mm_b2,
    float* __restrict__ sums, float* __restrict__ counts) {
  __shared__ __bf16 sA32[16][32];    // edge_attr zero-padded to K=32
  __shared__ __bf16 sHe[16][H128];   // edge MLP hidden (post-SiLU)
  __shared__ __bf16 sEF[16][H128];   // edge features
  __shared__ __bf16 sHN[16][H128];   // gathered h_nodes[src] (async copy dest)
  __shared__ __bf16 sM1[16][H128];   // message MLP hidden (post-SiLU)
  __shared__ int sSrc[16];
  __shared__ int sDst[16];

  const int tid   = threadIdx.x;
  const int wave  = tid >> 5;
  const int lane  = tid & 31;
  const int half  = lane >> 4;
  const int nn    = (lane & 15) + wave * 16;
  const int khalf = half * 8;
  const int e0    = blockIdx.x * 16;

  if (tid < 16) {
    int e = e0 + tid;
    if (e < E_total) {
      int s = edge_index[e];
      int d = edge_index[E_total + e];
      sSrc[tid] = s;
      sDst[tid] = d;
      float px = node_pos[s * 3 + 0], py = node_pos[s * 3 + 1], pz = node_pos[s * 3 + 2];
      float rx = grid_pos[d * 3 + 0] - px;
      float ry = grid_pos[d * 3 + 1] - py;
      float rz = grid_pos[d * 3 + 2] - pz;
      const float* R = orient + (size_t)s * 9;
      float t0 = rx * R[0] + ry * R[3] + rz * R[6];
      float t1 = rx * R[1] + ry * R[4] + rz * R[7];
      float t2 = rx * R[2] + ry * R[5] + rz * R[8];
      sA32[tid][0] = (__bf16)px; sA32[tid][1] = (__bf16)py; sA32[tid][2] = (__bf16)pz;
      sA32[tid][3] = (__bf16)t0; sA32[tid][4] = (__bf16)t1; sA32[tid][5] = (__bf16)t2;
      atomicAdd(&counts[d], 1.0f);
    } else {
      sSrc[tid] = 0;
      sDst[tid] = -1;
#pragma unroll
      for (int j = 0; j < 6; ++j) sA32[tid][j] = (__bf16)0.0f;
    }
#pragma unroll
    for (int j = 6; j < 32; ++j) sA32[tid][j] = (__bf16)0.0f;
  }
  __syncthreads();

  // Async gather of h_nodes[src] (bf16): 16 rows x 256B = 256 chunks of 16B,
  // one GLOBAL_LOAD_ASYNC_TO_LDS_B128 per thread, no VGPR round-trip.
  {
    int r = tid >> 4;          // row 0..15
    int c = tid & 15;          // 16B chunk 0..15
    const __bf16* gsrc = h_nodes + (size_t)sSrc[r] * H128 + c * 8;
    unsigned lds_off = (unsigned)(size_t)(void*)&sHN[r][c * 8];
    async_copy_b128(lds_off, gsrc);
    wait_async0();
  }
  __syncthreads();

  // Edge layer 1: [16,6(pad32)] @ [6,128] + b -> SiLU  (single WMMA per wave)
  {
    v8f acc = {};
    v16bf a = a_from_bf16row(&sA32[lane & 15][0], khalf);
    v16bf b = b_frag(em_w1p, wave, 1, 0, lane);
    acc = wmma_bf16(a, b, acc);
    const float bias = em_b1[nn];
#pragma unroll
    for (int r = 0; r < 8; ++r)
      sHe[r + half * 8][nn] = (__bf16)silu_f(acc[r] + bias);
  }
  __syncthreads();

  // Edge layer 2: [16,128] @ [128,128] + b -> edge feats
  {
    v8f acc = {};
#pragma unroll
    for (int ks = 0; ks < 4; ++ks) {
      v16bf a = a_from_bf16row(&sHe[lane & 15][0], ks * 32 + khalf);
      v16bf b = b_frag(em_w2p, wave, 4, ks, lane);
      acc = wmma_bf16(a, b, acc);
    }
    const float bias = em_b2[nn];
#pragma unroll
    for (int r = 0; r < 8; ++r)
      sEF[r + half * 8][nn] = (__bf16)(acc[r] + bias);
  }
  __syncthreads();

  // Message layer 1: concat([16,128],[16,128]) @ [256,128] + b -> SiLU
  {
    v8f acc = {};
#pragma unroll
    for (int ks = 0; ks < 8; ++ks) {
      const __bf16* row = (ks < 4) ? &sHN[lane & 15][0] : &sEF[lane & 15][0];
      v16bf a = a_from_bf16row(row, (ks & 3) * 32 + khalf);
      v16bf b = b_frag(mm_w1p, wave, 8, ks, lane);
      acc = wmma_bf16(a, b, acc);
    }
    const float bias = mm_b1[nn];
#pragma unroll
    for (int r = 0; r < 8; ++r)
      sM1[r + half * 8][nn] = (__bf16)silu_f(acc[r] + bias);
  }
  __syncthreads();

  // Message layer 2 + scatter-add straight from accumulators.
  {
    v8f acc = {};
#pragma unroll
    for (int ks = 0; ks < 4; ++ks) {
      v16bf a = a_from_bf16row(&sM1[lane & 15][0], ks * 32 + khalf);
      v16bf b = b_frag(mm_w2p, wave, 4, ks, lane);
      acc = wmma_bf16(a, b, acc);
    }
    const float bias = mm_b2[nn];
#pragma unroll
    for (int r = 0; r < 8; ++r) {
      int m = half * 8 + r;
      int d = sDst[m];
      if (d >= 0) atomicAdd(&sums[(size_t)d * H128 + nn], acc[r] + bias);
    }
  }
}

// ---------------------------------------------------------------------------
extern "C" void kernel_launch(void* const* d_in, const int* in_sizes, int n_in,
                              void* d_out, int out_size, void* d_ws, size_t ws_size,
                              hipStream_t stream) {
  (void)n_in; (void)out_size; (void)ws_size;
  const float* node_features = (const float*)d_in[0];
  const float* node_pos      = (const float*)d_in[1];
  const float* grid_pos      = (const float*)d_in[2];
  const int*   edge_index    = (const int*)d_in[3];
  const float* orient        = (const float*)d_in[4];
  const float* nm_w1 = (const float*)d_in[5];
  const float* nm_b1 = (const float*)d_in[6];
  const float* nm_w2 = (const float*)d_in[7];
  const float* nm_b2 = (const float*)d_in[8];
  const float* em_w1 = (const float*)d_in[9];
  const float* em_b1 = (const float*)d_in[10];
  const float* em_w2 = (const float*)d_in[11];
  const float* em_b2 = (const float*)d_in[12];
  const float* mm_w1 = (const float*)d_in[13];
  const float* mm_b1 = (const float*)d_in[14];
  const float* mm_w2 = (const float*)d_in[15];
  const float* mm_b2 = (const float*)d_in[16];
  const float* um_w1 = (const float*)d_in[17];
  const float* um_b1 = (const float*)d_in[18];
  const float* um_w2 = (const float*)d_in[19];
  const float* um_b2 = (const float*)d_in[20];

  const int N = in_sizes[0] / H128;   // 100000
  const int G = in_sizes[2] / 3;      // 32768
  const int E = in_sizes[3] / 2;      // 600000

  char* ws = (char*)d_ws;
  size_t off = 0;
  __bf16* h_nodes = (__bf16*)(ws + off); off += (size_t)N * H128 * sizeof(__bf16);
  off = (off + 255) & ~(size_t)255;
  float* sums    = (float*)(ws + off); off += (size_t)G * H128 * sizeof(float);
  float* counts  = (float*)(ws + off); off += (size_t)G * sizeof(float);
  __bf16* p_nm1 = (__bf16*)(ws + off); off += (size_t)128 * 128 * 2;
  __bf16* p_nm2 = (__bf16*)(ws + off); off += (size_t)128 * 128 * 2;
  __bf16* p_em1 = (__bf16*)(ws + off); off += (size_t)32  * 128 * 2;
  __bf16* p_em2 = (__bf16*)(ws + off); off += (size_t)128 * 128 * 2;
  __bf16* p_mm1 = (__bf16*)(ws + off); off += (size_t)256 * 128 * 2;
  __bf16* p_mm2 = (__bf16*)(ws + off); off += (size_t)128 * 128 * 2;
  __bf16* p_um1 = (__bf16*)(ws + off); off += (size_t)128 * 128 * 2;
  __bf16* p_um2 = (__bf16*)(ws + off); off += (size_t)128 * 128 * 2;

  // Re-zero sums+counts every call (contiguous region).
  long long nz = (long long)G * H128 + G;
  zero_f32_kernel<<<(int)((nz + 255) / 256), 256, 0, stream>>>(sums, nz);

  // Pack weights into WMMA B-fragment layout (bf16).
  pack_w_kernel<<<4, 256, 0, stream>>>(nm_w1, p_nm1, 4, 128);
  pack_w_kernel<<<4, 256, 0, stream>>>(nm_w2, p_nm2, 4, 128);
  pack_w_kernel<<<1, 256, 0, stream>>>(em_w1, p_em1, 1, 6);
  pack_w_kernel<<<4, 256, 0, stream>>>(em_w2, p_em2, 4, 128);
  pack_w_kernel<<<8, 256, 0, stream>>>(mm_w1, p_mm1, 8, 256);
  pack_w_kernel<<<4, 256, 0, stream>>>(mm_w2, p_mm2, 4, 128);
  pack_w_kernel<<<4, 256, 0, stream>>>(um_w1, p_um1, 4, 128);
  pack_w_kernel<<<4, 256, 0, stream>>>(um_w2, p_um2, 4, 128);

  // Node MLP: h_nodes (bf16) = mlp(node_features)
  mlp128_kernel<<<(N + 15) / 16, 256, 0, stream>>>(
      node_features, nullptr, p_nm1, nm_b1, p_nm2, nm_b2,
      nullptr, h_nodes, N);

  // Edge + message MLPs, scatter-add into sums/counts.
  edge_msg_kernel<<<(E + 15) / 16, 256, 0, stream>>>(
      h_nodes, node_pos, grid_pos, edge_index, E, orient,
      p_em1, em_b1, p_em2, em_b2, p_mm1, mm_b1, p_mm2, mm_b2, sums, counts);

  // Update MLP on grid_feats = sums / max(counts,1) -> d_out (f32).
  mlp128_kernel<<<(G + 15) / 16, 256, 0, stream>>>(
      sums, counts, p_um1, um_b1, p_um2, um_b2,
      (float*)d_out, nullptr, G);
}
